// EfficientAttention_44358422233457
// MI455X (gfx1250) — matrix operand order repaired and checked by reading, not compiled, not run on hardware
//
#include <hip/hip_runtime.h>

#define BDIM 8
#define CDIM 768
#define SDIM 1024
#define HEADS 12
#define HD 64

typedef __bf16 bf16_t;
typedef __attribute__((ext_vector_type(16))) bf16_t v16bf;
typedef __attribute__((ext_vector_type(8))) float v8f;
typedef __attribute__((ext_vector_type(4))) unsigned int u32x4;

struct ABReg {
  union {
    v16bf v;
    u32x4 u[2];
  };
};

#define WMMA_BF16(A, B, Cacc) \
  __builtin_amdgcn_wmma_f32_16x16x32_bf16(false, (A), false, (B), (short)0, (Cacc), false, false)

__device__ __forceinline__ unsigned short f2bf(float f) {
  unsigned int u = __builtin_bit_cast(unsigned int, f);
  u += 0x7FFFu + ((u >> 16) & 1u);  // round-to-nearest-even
  return (unsigned short)(u >> 16);
}

// Generic shared pointer -> LDS byte offset (low 32 bits of the generic address).
__device__ __forceinline__ unsigned lds_off(const void* p) {
  return (unsigned)(size_t)p;
}

// CDNA5 async DMA: copy 16 bytes global -> LDS, tracked by ASYNCcnt (GV mode).
__device__ __forceinline__ void async_ld16(unsigned lds, const void* gptr) {
  asm volatile("global_load_async_to_lds_b128 %0, %1, off"
               :: "v"(lds), "v"((unsigned long long)(size_t)gptr)
               : "memory");
}

__device__ __forceinline__ void wait_async0() {
  asm volatile("s_wait_asynccnt 0" ::: "memory");
}

// A-matrix 16x32 bf16 from row-major memory (row stride ld elements).
// lanes 0-15: row m, K 0..7 (v0-3) & 16..23 (v4-7); lanes 16-31: K 8..15 & 24..31.
__device__ __forceinline__ ABReg load_a_g(const unsigned short* p, int ld, int row0, int k0) {
  const int lane = threadIdx.x & 31;
  const unsigned short* base = p + (size_t)(row0 + (lane & 15)) * ld + k0 + (lane >> 4) * 8;
  ABReg r;
  r.u[0] = *(const u32x4*)(base);
  r.u[1] = *(const u32x4*)(base + 16);
  return r;
}

// B-matrix 32x16 bf16 from LDS where column n is contiguous at p + n*ld.
// lane n (0-15): column n, K 0..7 & 16..23; lane n+16: column n, K 8..15 & 24..31.
__device__ __forceinline__ ABReg load_b_lds(const unsigned short* p, int ld, int col0, int k0) {
  const int lane = threadIdx.x & 31;
  const unsigned short* base = p + (size_t)(col0 + (lane & 15)) * ld + k0 + (lane >> 4) * 8;
  ABReg r;
  r.u[0] = *(const u32x4*)(base);
  r.u[1] = *(const u32x4*)(base + 16);
  return r;
}

// ---------------- Kernel 0: transpose (C,S) -> (S,C) per batch, fp32 -> bf16
__global__ __launch_bounds__(256) void k_transpose_cvt(const float* __restrict__ src,
                                                       unsigned short* __restrict__ dst) {
  __shared__ float tile[32][33];
  const int b = blockIdx.z;
  const int c0 = blockIdx.y * 32;
  const int s0 = blockIdx.x * 32;
  const int tx = threadIdx.x, ty = threadIdx.y;  // 32 x 8
  const float* s = src + (size_t)b * CDIM * SDIM;
  unsigned short* d = dst + (size_t)b * SDIM * CDIM;
#pragma unroll
  for (int i = 0; i < 4; ++i)
    tile[ty + i * 8][tx] = s[(size_t)(c0 + ty + i * 8) * SDIM + s0 + tx];
  __syncthreads();
#pragma unroll
  for (int i = 0; i < 4; ++i)
    d[(size_t)(s0 + ty + i * 8) * CDIM + c0 + tx] = f2bf(tile[tx][ty + i * 8]);
}

// ---------------- Kernel 1: flat fp32 -> bf16 convert (weights)
__global__ __launch_bounds__(256) void k_cvt(const float* __restrict__ src,
                                             unsigned short* __restrict__ dst, int n) {
  int i = blockIdx.x * 256 + threadIdx.x;
  if (i < n) dst[i] = f2bf(src[i]);
}

// ---------------- Kernel 2: projection GEMM  Y[s,o] = sum_c xT[s,c]*W[o,c] + bias[o]
// Weight tile (64 o x 64 c, 8KB) async-DMA'd to LDS; two buffers, manual x2 unroll
// so accumulators stay pinned. mode 0: store (B,H,S,D).  mode 1: store (B,H,D,S).
__global__ __launch_bounds__(256) void k_proj(const unsigned short* __restrict__ xT,
                                              const unsigned short* __restrict__ W,
                                              const float* __restrict__ bias,
                                              unsigned short* __restrict__ out, int mode) {
  __shared__ __align__(16) unsigned short wt0[64 * 64];
  __shared__ __align__(16) unsigned short wt1[64 * 64];
  const int tid = threadIdx.x;
  const int b = blockIdx.z;
  const int o0 = blockIdx.x * 64;
  const int wave = tid >> 5;
  const int lane = tid & 31;
  const int half = lane >> 4;
  const int srow = blockIdx.y * 128 + wave * 16;
  const unsigned short* x = xT + (size_t)b * SDIM * CDIM;

  v8f acc[4] = {};
  auto stage = [&](unsigned short* buf, int c) {
#pragma unroll
    for (int i = 0; i < 2; ++i) {
      const int ch = tid + i * 256;            // 512 chunks of 16B
      const int row = ch >> 3, off = (ch & 7) * 8;
      async_ld16(lds_off(buf + row * 64 + off),
                 W + (size_t)(o0 + row) * CDIM + c + off);
    }
  };
  auto body = [&](const unsigned short* buf, int c) {
#pragma unroll
    for (int cc = 0; cc < 64; cc += 32) {
      ABReg a = load_a_g(x, CDIM, srow, c + cc);
#pragma unroll
      for (int j = 0; j < 4; ++j) {
        ABReg bw = load_b_lds(buf, 64, j * 16, cc);
        acc[j] = WMMA_BF16(a.v, bw.v, acc[j]);
      }
    }
  };

  stage(wt0, 0);
  for (int c = 0; c < CDIM; c += 128) {
    wait_async0();
    __syncthreads();
    if (c + 64 < CDIM) stage(wt1, c + 64);
    __builtin_prefetch(x + (size_t)(srow + (lane & 15)) * CDIM + c + 128, 0, 1);
    body(wt0, c);
    __syncthreads();

    wait_async0();
    __syncthreads();
    if (c + 128 < CDIM) stage(wt0, c + 128);
    body(wt1, c + 64);
    __syncthreads();
  }

#pragma unroll
  for (int j = 0; j < 4; ++j) {
    const int og = o0 + j * 16 + (lane & 15);
    const float bval = bias[og];
    const int head = og >> 6, dc = og & 63;
#pragma unroll
    for (int r = 0; r < 8; ++r) {
      const int sg = srow + r + 8 * half;
      const unsigned short val = f2bf(acc[j][r] + bval);
      if (mode == 0)
        out[(((size_t)b * HEADS + head) * SDIM + sg) * HD + dc] = val;
      else
        out[(((size_t)b * HEADS + head) * HD + dc) * SDIM + sg] = val;
    }
  }
}

// ---------------- Kernel 3: flash attention, one 16-row q tile per wave.
// K tile (64 kpos x 64 d) and V tile (64 d x 64 s) async-DMA'd to LDS,
// two buffers, manual x2 unroll; shared by all 8 waves of the block.
__global__ __launch_bounds__(256) void k_attn(const unsigned short* __restrict__ qe,
                                              const unsigned short* __restrict__ ke,
                                              const unsigned short* __restrict__ veT,
                                              unsigned short* __restrict__ svb) {
  __shared__ __align__(16) unsigned short kl0[64 * 64];
  __shared__ __align__(16) unsigned short kl1[64 * 64];
  __shared__ __align__(16) unsigned short vl0[64 * 64];
  __shared__ __align__(16) unsigned short vl1[64 * 64];
  __shared__ __align__(16) unsigned short plds[8][16 * 32];  // per-wave P tile
  const int tid = threadIdx.x;
  const int bh = blockIdx.y;
  const int b = bh / HEADS, h = bh % HEADS;
  const int wave = tid >> 5;
  const int lane = tid & 31;
  const int half = lane >> 4;
  const int col = lane & 15;
  const int q0 = (blockIdx.x * 8 + wave) * 16;
  const unsigned short* qh = qe + (size_t)bh * SDIM * HD;
  const unsigned short* kh = ke + (size_t)bh * SDIM * HD;
  const unsigned short* vh = veT + (size_t)bh * HD * SDIM;

  ABReg aq0 = load_a_g(qh, HD, q0, 0);
  ABReg aq1 = load_a_g(qh, HD, q0, 32);
  v8f acc[4] = {};
  float mrow[8], lrow[8];
#pragma unroll
  for (int r = 0; r < 8; ++r) { mrow[r] = -1e30f; lrow[r] = 0.f; }
  const float sm = 0.125f;  // 1/sqrt(64)

  auto stage = [&](unsigned short* kbuf, unsigned short* vbuf, int kb) {
#pragma unroll
    for (int i = 0; i < 2; ++i) {
      const int ch = tid + i * 256;            // 512 chunks of 16B per tile
      const int row = ch >> 3, off = (ch & 7) * 8;
      async_ld16(lds_off(kbuf + row * 64 + off),
                 kh + (size_t)(kb + row) * HD + off);      // [kpos][d]
      async_ld16(lds_off(vbuf + row * 64 + off),
                 vh + (size_t)row * SDIM + kb + off);      // [d][s]
    }
  };

  auto chunk = [&](const unsigned short* kbuf, const unsigned short* vbuf) {
#pragma unroll
    for (int kk = 0; kk < 64; kk += 32) {
      // QK^T : two 16x16 score tiles; B columns = kpos (d contiguous in kbuf rows)
      ABReg b00 = load_b_lds(kbuf, 64, kk, 0);
      ABReg b01 = load_b_lds(kbuf, 64, kk, 32);
      ABReg b10 = load_b_lds(kbuf, 64, kk + 16, 0);
      ABReg b11 = load_b_lds(kbuf, 64, kk + 16, 32);
      v8f s0 = {}, s1 = {};
      s0 = WMMA_BF16(aq0.v, b00.v, s0);
      s0 = WMMA_BF16(aq1.v, b01.v, s0);
      s1 = WMMA_BF16(aq0.v, b10.v, s1);
      s1 = WMMA_BF16(aq1.v, b11.v, s1);

      // online softmax; D layout: VGPR r <-> row r + 8*half, col = lane&15
#pragma unroll
      for (int r = 0; r < 8; ++r) {
        float a0 = s0[r] * sm, a1 = s1[r] * sm;
        float t = fmaxf(a0, a1);
#pragma unroll
        for (int off = 1; off < 16; off <<= 1) t = fmaxf(t, __shfl_xor(t, off));
        const float mnew = fmaxf(mrow[r], t);
        const float sc = __expf(mrow[r] - mnew);
        const float p0 = __expf(a0 - mnew);
        const float p1 = __expf(a1 - mnew);
        float ps = p0 + p1;
#pragma unroll
        for (int off = 1; off < 16; off <<= 1) ps += __shfl_xor(ps, off);
        lrow[r] = lrow[r] * sc + ps;
        mrow[r] = mnew;
        acc[0][r] *= sc;
        acc[1][r] *= sc;
        acc[2][r] *= sc;
        acc[3][r] *= sc;
        const int rr = r + 8 * half;
        plds[wave][rr * 32 + col] = f2bf(p0);
        plds[wave][rr * 32 + col + 16] = f2bf(p1);
      }
      asm volatile("s_wait_dscnt 0" ::: "memory");
      // reload probabilities in A layout (same wave; DS ops are in-order)
      ABReg ap;
      {
        const unsigned short* pb = &plds[wave][(lane & 15) * 32 + half * 8];
        ap.u[0] = *(const u32x4*)(pb);
        ap.u[1] = *(const u32x4*)(pb + 16);
      }
      // P x V : B columns = d (s contiguous in vbuf rows)
#pragma unroll
      for (int j = 0; j < 4; ++j) {
        ABReg bvt = load_b_lds(vbuf, 64, j * 16, kk);
        acc[j] = WMMA_BF16(ap.v, bvt.v, acc[j]);
      }
    }
  };

  stage(kl0, vl0, 0);
  for (int kb = 0; kb < SDIM; kb += 128) {
    wait_async0();
    __syncthreads();
    if (kb + 64 < SDIM) stage(kl1, vl1, kb + 64);
    chunk(kl0, vl0);
    __syncthreads();

    wait_async0();
    __syncthreads();
    if (kb + 128 < SDIM) stage(kl0, vl0, kb + 128);
    chunk(kl1, vl1);
    __syncthreads();
  }

  // epilogue: normalize and store sv as (B,S,C) bf16
  unsigned short* so = svb + (size_t)b * SDIM * CDIM + (size_t)h * HD;
#pragma unroll
  for (int r = 0; r < 8; ++r) {
    const float inv = 1.0f / lrow[r];
    const int sg = q0 + r + 8 * half;
#pragma unroll
    for (int j = 0; j < 4; ++j)
      so[(size_t)sg * CDIM + j * 16 + col] = f2bf(acc[j][r] * inv);
  }
}

// ---------------- Kernel 4: output projection  out[b,o,s] = sum_c Wo[o,c]*sv[b,s,c] + bo[o]
// sv tile (64 s x 64 c, 8KB) async-DMA'd to LDS, two buffers, manual x2 unroll.
__global__ __launch_bounds__(256) void k_outproj(const unsigned short* __restrict__ Wob,
                                                 const unsigned short* __restrict__ svb,
                                                 const float* __restrict__ bo,
                                                 float* __restrict__ out) {
  __shared__ __align__(16) unsigned short st0[64 * 64];
  __shared__ __align__(16) unsigned short st1[64 * 64];
  const int tid = threadIdx.x;
  const int b = blockIdx.z;
  const int wave = tid >> 5;
  const int lane = tid & 31;
  const int half = lane >> 4;
  const int o0 = blockIdx.y * 128 + wave * 16;
  const int s0 = blockIdx.x * 64;
  const unsigned short* sv = svb + (size_t)b * SDIM * CDIM;

  v8f acc[4] = {};
  auto stage = [&](unsigned short* buf, int c) {
#pragma unroll
    for (int i = 0; i < 2; ++i) {
      const int ch = tid + i * 256;
      const int row = ch >> 3, off = (ch & 7) * 8;
      async_ld16(lds_off(buf + row * 64 + off),
                 sv + (size_t)(s0 + row) * CDIM + c + off);
    }
  };
  auto body = [&](const unsigned short* buf, int c) {
#pragma unroll
    for (int cc = 0; cc < 64; cc += 32) {
      ABReg a = load_a_g(Wob, CDIM, o0, c + cc);
#pragma unroll
      for (int j = 0; j < 4; ++j) {
        ABReg bt = load_b_lds(buf, 64, j * 16, cc);
        acc[j] = WMMA_BF16(a.v, bt.v, acc[j]);
      }
    }
  };

  stage(st0, 0);
  for (int c = 0; c < CDIM; c += 128) {
    wait_async0();
    __syncthreads();
    if (c + 64 < CDIM) stage(st1, c + 64);
    __builtin_prefetch(Wob + (size_t)(o0 + (lane & 15)) * CDIM + c + 128, 0, 1);
    body(st0, c);
    __syncthreads();

    wait_async0();
    __syncthreads();
    if (c + 128 < CDIM) stage(st0, c + 128);
    body(st1, c + 64);
    __syncthreads();
  }

  float bvv[8];
#pragma unroll
  for (int r = 0; r < 8; ++r) bvv[r] = bo[o0 + r + 8 * half];
#pragma unroll
  for (int j = 0; j < 4; ++j) {
    const int sg = s0 + j * 16 + (lane & 15);
#pragma unroll
    for (int r = 0; r < 8; ++r) {
      const int og = o0 + r + 8 * half;
      out[((size_t)b * CDIM + og) * SDIM + sg] = acc[j][r] + bvv[r];
    }
  }
}

extern "C" void kernel_launch(void* const* d_in, const int* in_sizes, int n_in,
                              void* d_out, int out_size, void* d_ws, size_t ws_size,
                              hipStream_t stream) {
  (void)in_sizes; (void)n_in; (void)out_size; (void)ws_size;
  const float* q  = (const float*)d_in[0];
  const float* k  = (const float*)d_in[1];
  const float* v  = (const float*)d_in[2];
  const float* Wq = (const float*)d_in[3];
  const float* bq = (const float*)d_in[4];
  const float* Wk = (const float*)d_in[5];
  const float* bk = (const float*)d_in[6];
  const float* Wv = (const float*)d_in[7];
  const float* bv = (const float*)d_in[8];
  const float* Wo = (const float*)d_in[9];
  const float* bo = (const float*)d_in[10];
  float* out = (float*)d_out;

  unsigned short* ws = (unsigned short*)d_ws;
  const size_t SC = (size_t)SDIM * CDIM;
  const size_t CC = (size_t)CDIM * CDIM;
  unsigned short* qT  = ws;
  unsigned short* kT  = qT + BDIM * SC;
  unsigned short* vT  = kT + BDIM * SC;
  unsigned short* Wqb = vT + BDIM * SC;
  unsigned short* Wkb = Wqb + CC;
  unsigned short* Wvb = Wkb + CC;
  unsigned short* Wob = Wvb + CC;
  unsigned short* qe  = Wob + CC;
  unsigned short* ke  = qe + BDIM * SC;
  unsigned short* veT = ke + BDIM * SC;
  unsigned short* svb = veT + BDIM * SC;

  // 0) transpose + convert activations
  dim3 tb(32, 8, 1);
  dim3 tg(SDIM / 32, CDIM / 32, BDIM);
  k_transpose_cvt<<<tg, tb, 0, stream>>>(q, qT);
  k_transpose_cvt<<<tg, tb, 0, stream>>>(k, kT);
  k_transpose_cvt<<<tg, tb, 0, stream>>>(v, vT);

  // 1) convert weights
  const int n = CDIM * CDIM;
  k_cvt<<<(n + 255) / 256, 256, 0, stream>>>(Wq, Wqb, n);
  k_cvt<<<(n + 255) / 256, 256, 0, stream>>>(Wk, Wkb, n);
  k_cvt<<<(n + 255) / 256, 256, 0, stream>>>(Wv, Wvb, n);
  k_cvt<<<(n + 255) / 256, 256, 0, stream>>>(Wo, Wob, n);

  // 2) projections
  dim3 pg(CDIM / 64, SDIM / 128, BDIM);  // (12, 8, 8)
  k_proj<<<pg, 256, 0, stream>>>(qT, Wqb, bq, qe, 0);
  k_proj<<<pg, 256, 0, stream>>>(kT, Wkb, bk, ke, 0);
  k_proj<<<pg, 256, 0, stream>>>(vT, Wvb, bv, veT, 1);

  // 3) attention
  dim3 ag(SDIM / 128, BDIM * HEADS, 1);  // (8, 96)
  k_attn<<<ag, 256, 0, stream>>>(qe, ke, veT, svb);

  // 4) output projection
  dim3 og(SDIM / 64, CDIM / 128, BDIM);  // (16, 6, 8)
  k_outproj<<<og, 256, 0, stream>>>(Wob, svb, bo, out);
}
